// NeuralEmbeddingLayer_80152679678607
// MI455X (gfx1250) — matrix operand order repaired and checked by reading, not compiled, not run on hardware
//
#include <hip/hip_runtime.h>
#include <hip/hip_bf16.h>

// ---------------- problem constants ----------------
#define B_    32
#define T_    1024
#define C_    256
#define D_    512
#define H_    1024
#define KWIN  32
#define SSTR  4
#define L_    249           // (T - K)/S + 1
#define KD_   (KWIN * D_)   // 16384
#define TOK_  (L_ + 2)      // 251 tokens per batch in output
#define NDAYS 24

typedef __bf16 v16bf __attribute__((ext_vector_type(16)));
typedef float  v8f   __attribute__((ext_vector_type(8)));

union FragAB { unsigned int u[8]; v16bf v; };

// fp32 -> bf16 (scalar, RNE)
__device__ __forceinline__ unsigned short cvt_bf16(float a) {
    unsigned int u = __float_as_uint(a);
    u = (u + 0x7FFFu + ((u >> 16) & 1u)) >> 16;
    return (unsigned short)u;
}

// two fp32 -> packed bf16x2: RNE-round + one v_perm_b32 packing the high halves
__device__ __forceinline__ unsigned int pack_bf16x2(float a, float b) {
    unsigned int ua = __float_as_uint(a), ub = __float_as_uint(b);
    ua += 0x7FFFu + ((ua >> 16) & 1u);
    ub += 0x7FFFu + ((ub >> 16) & 1u);
    return __builtin_amdgcn_perm(ub, ua, 0x07060302u);  // {b.hi16, a.hi16}
}

// =====================================================================
// Prep: bulk fp32 -> bf16 conversion (grid-stride, float4 granularity)
// =====================================================================
__global__ __launch_bounds__(256)
void k_cvt_bf16(const float* __restrict__ src,
                unsigned short* __restrict__ dst, int n4)
{
    int i = blockIdx.x * 256 + threadIdx.x;
    const int stride = gridDim.x * 256;
    for (; i < n4; i += stride) {
        const float4 f = ((const float4*)src)[i];
        ((uint2*)dst)[i] = make_uint2(pack_bf16x2(f.x, f.y), pack_bf16x2(f.z, f.w));
    }
}

// =====================================================================
// Kernel 1: x[b,t,d] = GELU(spikes[b] @ W_spikes[day[b]]^T + b_spikes[day[b]])
// bf16 output to workspace. M=T, N=D, K=C. WG 64x128, 8 waves of 32x32.
// =====================================================================
__global__ __launch_bounds__(256)
void k_spike_gemm(const float* __restrict__ spikes,
                  const unsigned short* __restrict__ WspBf,
                  const float* __restrict__ bsp,
                  const int*   __restrict__ day_idx,
                  unsigned short* __restrict__ xbf)
{
    __shared__ __align__(16) unsigned short sA[64 * 72];
    __shared__ __align__(16) unsigned short sB[128 * 72];

    const int b   = blockIdx.z;
    const int t0  = blockIdx.x * 64;
    const int n0  = blockIdx.y * 128;
    const int tid = threadIdx.x;
    const int lane = tid & 31, wave = tid >> 5;
    const int wm = wave >> 2, wn = wave & 3;          // 2 x 4 wave grid
    const int half = lane >> 4, r16 = lane & 15;

    const int day = day_idx[b];
    const unsigned short* Wd = WspBf + (size_t)day * D_ * C_;
    const float* Ab = spikes + (size_t)b * T_ * C_;

    const int nS = tid >> 1, hkS = tid & 1;

    v8f z = {0.f,0.f,0.f,0.f,0.f,0.f,0.f,0.f};
    v8f acc[2][2];
    acc[0][0]=z; acc[0][1]=z; acc[1][0]=z; acc[1][1]=z;

    float4 aF[4];
    uint4  bR[4];

    // preload k-chunk 0
    #pragma unroll
    for (int i = 0; i < 4; ++i) {
        int c = tid + i * 256, row = c >> 4, q = c & 15;
        aF[i] = *(const float4*)(Ab + (size_t)(t0 + row) * C_ + q * 4);
    }
    #pragma unroll
    for (int j = 0; j < 4; ++j)
        bR[j] = *(const uint4*)(Wd + (size_t)(n0 + nS) * C_ + hkS * 32 + j * 8);

    for (int it = 0; it < C_ / 64; ++it) {
        // commit registers -> LDS
        #pragma unroll
        for (int i = 0; i < 4; ++i) {
            int c = tid + i * 256, row = c >> 4, q = c & 15;
            *(uint2*)&sA[row * 72 + q * 4] =
                make_uint2(pack_bf16x2(aF[i].x, aF[i].y), pack_bf16x2(aF[i].z, aF[i].w));
        }
        #pragma unroll
        for (int j = 0; j < 4; ++j)
            *(uint4*)&sB[nS * 72 + hkS * 32 + j * 8] = bR[j];
        __syncthreads();

        // issue next tile's global loads (in flight during WMMAs)
        if (it + 1 < C_ / 64) {
            const int c0 = (it + 1) * 64;
            #pragma unroll
            for (int i = 0; i < 4; ++i) {
                int c = tid + i * 256, row = c >> 4, q = c & 15;
                aF[i] = *(const float4*)(Ab + (size_t)(t0 + row) * C_ + c0 + q * 4);
            }
            #pragma unroll
            for (int j = 0; j < 4; ++j)
                bR[j] = *(const uint4*)(Wd + (size_t)(n0 + nS) * C_ + c0 + hkS * 32 + j * 8);
        }

        // consume LDS tile: 2 k-steps of 32
        #pragma unroll
        for (int kk = 0; kk < 64; kk += 32) {
            FragAB a[2], bb[2];
            #pragma unroll
            for (int mi = 0; mi < 2; ++mi) {
                const int row = wm * 32 + mi * 16 + r16;
                #pragma unroll
                for (int j = 0; j < 8; ++j) {
                    const int k0 = ((j & 3) << 1) + ((j >> 2) << 4) + (half << 3);
                    a[mi].u[j] = *(const unsigned int*)&sA[row * 72 + kk + k0];
                }
            }
            #pragma unroll
            for (int ni = 0; ni < 2; ++ni) {
                const int col = wn * 32 + ni * 16 + r16;
                #pragma unroll
                for (int j = 0; j < 8; ++j) {
                    const int k0 = ((j & 3) << 1) + ((j >> 2) << 4) + (half << 3);
                    bb[ni].u[j] = *(const unsigned int*)&sB[col * 72 + kk + k0];
                }
            }
            #pragma unroll
            for (int mi = 0; mi < 2; ++mi)
                #pragma unroll
                for (int ni = 0; ni < 2; ++ni)
                    acc[mi][ni] = __builtin_amdgcn_wmma_f32_16x16x32_bf16(
                        false, a[mi].v, false, bb[ni].v,
                        (short)0, acc[mi][ni], false, false);
        }
        __syncthreads();
    }

    // epilogue: bias + exact GELU, store bf16
    #pragma unroll
    for (int mi = 0; mi < 2; ++mi)
        #pragma unroll
        for (int ni = 0; ni < 2; ++ni)
            #pragma unroll
            for (int r = 0; r < 8; ++r) {
                const int t = t0 + wm * 32 + mi * 16 + r + half * 8;
                const int d = n0 + wn * 32 + ni * 16 + r16;
                float v = acc[mi][ni][r] + bsp[day * D_ + d];
                v = 0.5f * v * (1.0f + erff(v * 0.70710678118654752f));
                xbf[(size_t)(b * T_ + t) * D_ + d] = cvt_bf16(v);
            }
}

// =====================================================================
// Kernel 2: out[b,l+2,h] = window(x) @ W_proj^T + b_proj + pos_table[ts]
// Pure-copy staging (both operands bf16). WG tile 128(M) x 128(N),
// 8 waves in 4x2, each wave 32x64 = 2x4 WMMA accumulators
// (12 ds_load_b128 per 8 WMMAs). K=16384 in chunks of 64.
// =====================================================================
__global__ __launch_bounds__(256)
void k_proj_gemm(const unsigned short* __restrict__ xbf,
                 const unsigned short* __restrict__ WpBf,
                 const float* __restrict__ bproj,
                 const float* __restrict__ pos_table,
                 const int*   __restrict__ tstamp,
                 float* __restrict__ out)
{
    __shared__ __align__(16) unsigned short sA[128 * 72];
    __shared__ __align__(16) unsigned short sB[128 * 72];

    const int b   = blockIdx.z;
    const int l0  = blockIdx.x * 128;
    const int h0  = blockIdx.y * 128;
    const int tid = threadIdx.x;
    const int lane = tid & 31, wave = tid >> 5;
    const int wm = wave >> 1, wn = wave & 1;          // 4 x 2 wave grid
    const int half = lane >> 4, r16 = lane & 15;

    const int nS = tid >> 1, hkS = tid & 1;           // B staging coords

    v8f z = {0.f,0.f,0.f,0.f,0.f,0.f,0.f,0.f};
    v8f acc[2][4];
    #pragma unroll
    for (int mi = 0; mi < 2; ++mi)
        #pragma unroll
        for (int ni = 0; ni < 4; ++ni) acc[mi][ni] = z;

    uint4 aR[4], bR[4];
    const uint4 zero4 = make_uint4(0u, 0u, 0u, 0u);

    // preload k-chunk 0
    #pragma unroll
    for (int i = 0; i < 4; ++i) {
        int c = tid + i * 256, row = c >> 3, q = c & 7;
        const int l = l0 + row;
        aR[i] = (l < L_)
            ? *(const uint4*)(xbf + (size_t)(b * T_ + l * SSTR) * D_ + q * 8)
            : zero4;
    }
    #pragma unroll
    for (int j = 0; j < 4; ++j)
        bR[j] = *(const uint4*)(WpBf + (size_t)(h0 + nS) * KD_ + hkS * 32 + j * 8);

    for (int it = 0; it < KD_ / 64; ++it) {
        // commit registers -> LDS
        #pragma unroll
        for (int i = 0; i < 4; ++i) {
            int c = tid + i * 256, row = c >> 3, q = c & 7;
            *(uint4*)&sA[row * 72 + q * 8] = aR[i];
        }
        #pragma unroll
        for (int j = 0; j < 4; ++j)
            *(uint4*)&sB[nS * 72 + hkS * 32 + j * 8] = bR[j];
        __syncthreads();

        // issue next tile's global loads (overlap with WMMAs)
        if (it + 1 < KD_ / 64) {
            const int kbase = (it + 1) * 64;
            const int kw = kbase >> 9;       // window tap (0..31)
            const int d0 = kbase & 511;      // d-slice within x row
            #pragma unroll
            for (int i = 0; i < 4; ++i) {
                int c = tid + i * 256, row = c >> 3, q = c & 7;
                const int l = l0 + row;
                aR[i] = (l < L_)
                    ? *(const uint4*)(xbf + (size_t)(b * T_ + l * SSTR + kw) * D_ + d0 + q * 8)
                    : zero4;
            }
            #pragma unroll
            for (int j = 0; j < 4; ++j)
                bR[j] = *(const uint4*)(WpBf + (size_t)(h0 + nS) * KD_ + kbase + hkS * 32 + j * 8);
        }

        // consume LDS tile: 2 k-steps of 32
        #pragma unroll
        for (int kk = 0; kk < 64; kk += 32) {
            FragAB a[2], bb[4];
            #pragma unroll
            for (int mi = 0; mi < 2; ++mi) {
                const int row = wm * 32 + mi * 16 + r16;
                #pragma unroll
                for (int j = 0; j < 8; ++j) {
                    const int k0 = ((j & 3) << 1) + ((j >> 2) << 4) + (half << 3);
                    a[mi].u[j] = *(const unsigned int*)&sA[row * 72 + kk + k0];
                }
            }
            #pragma unroll
            for (int ni = 0; ni < 4; ++ni) {
                const int col = wn * 64 + ni * 16 + r16;
                #pragma unroll
                for (int j = 0; j < 8; ++j) {
                    const int k0 = ((j & 3) << 1) + ((j >> 2) << 4) + (half << 3);
                    bb[ni].u[j] = *(const unsigned int*)&sB[col * 72 + kk + k0];
                }
            }
            #pragma unroll
            for (int mi = 0; mi < 2; ++mi)
                #pragma unroll
                for (int ni = 0; ni < 4; ++ni)
                    acc[mi][ni] = __builtin_amdgcn_wmma_f32_16x16x32_bf16(
                        false, a[mi].v, false, bb[ni].v,
                        (short)0, acc[mi][ni], false, false);
        }
        __syncthreads();
    }

    // epilogue: + b_proj + pos_table[timestamp], write token l+2
    #pragma unroll
    for (int mi = 0; mi < 2; ++mi)
        #pragma unroll
        for (int r = 0; r < 8; ++r) {
            const int l = l0 + wm * 32 + mi * 16 + r + half * 8;
            if (l < L_) {
                const int ts = tstamp[b * T_ + l];
                const float* posRow = pos_table + (size_t)ts * H_;
                float* outRow = out + ((size_t)b * TOK_ + (l + 2)) * H_;
                #pragma unroll
                for (int ni = 0; ni < 4; ++ni) {
                    const int h = h0 + wn * 64 + ni * 16 + r16;
                    outRow[h] = acc[mi][ni][r] + bproj[h] + posRow[h];
                }
            }
        }
}

// =====================================================================
// Kernel 3: day/block tokens, mask AND-reduction, timestamp copy
// Output layout: [x : B*TOK*H][mask : B*TOK][ts : B*L]
// =====================================================================
__global__ __launch_bounds__(256)
void k_epilogue(const float* __restrict__ block_table,
                const float* __restrict__ day_table,
                const int*   __restrict__ spikes_mask,
                const int*   __restrict__ tstamp,
                const int*   __restrict__ block_idx,
                const int*   __restrict__ day_idx,
                float* __restrict__ out)
{
    const int b   = blockIdx.x;
    const int tid = threadIdx.x;
    const int day = day_idx[b];
    const int blk = block_idx[b];

    float* xb = out + (size_t)b * TOK_ * H_;
    for (int h = tid; h < H_; h += 256) {
        xb[h]      = day_table[day * H_ + h];    // token 0
        xb[H_ + h] = block_table[blk * H_ + h];  // token 1
    }

    float* maskOut = out + (size_t)B_ * TOK_ * H_;
    float* tsOut   = maskOut + (size_t)B_ * TOK_;
    if (tid < 2) maskOut[b * TOK_ + tid] = 1.0f;
    for (int l = tid; l < L_; l += 256) {
        int m = 1;
        #pragma unroll
        for (int k = 0; k < KWIN; ++k)
            m *= spikes_mask[b * T_ + l * SSTR + k];
        maskOut[b * TOK_ + 2 + l] = (float)m;
        tsOut[b * L_ + l]         = (float)tstamp[b * T_ + l];
    }
}

// =====================================================================
extern "C" void kernel_launch(void* const* d_in, const int* in_sizes, int n_in,
                              void* d_out, int out_size, void* d_ws, size_t ws_size,
                              hipStream_t stream) {
    (void)in_sizes; (void)n_in; (void)out_size; (void)ws_size;
    const float* spikes      = (const float*)d_in[0];
    const float* W_spikes    = (const float*)d_in[1];
    const float* b_spikes    = (const float*)d_in[2];
    const float* W_proj      = (const float*)d_in[3];
    const float* b_proj      = (const float*)d_in[4];
    const float* pos_table   = (const float*)d_in[5];
    const float* block_table = (const float*)d_in[6];
    const float* day_table   = (const float*)d_in[7];
    const int*   spikes_mask = (const int*)d_in[8];
    const int*   tstamp      = (const int*)d_in[9];
    const int*   block_idx   = (const int*)d_in[10];
    const int*   day_idx     = (const int*)d_in[11];

    float* out = (float*)d_out;

    // workspace layout:
    //   [0,          32M)  x bf16           B*T*D
    //   [32M,        64M)  W_proj bf16      H*KD
    //   [64M,      70.3M)  W_spikes bf16    NDAYS*D*C
    unsigned short* xbf   = (unsigned short*)d_ws;
    unsigned short* WpBf  = xbf  + (size_t)B_ * T_ * D_;
    unsigned short* WspBf = WpBf + (size_t)H_ * KD_;

    // one-time weight conversions (bf16)
    k_cvt_bf16<<<dim3(2048), 256, 0, stream>>>(W_proj,   WpBf,  (H_ * KD_) / 4);
    k_cvt_bf16<<<dim3(512),  256, 0, stream>>>(W_spikes, WspBf, (NDAYS * D_ * C_) / 4);

    // GEMM1 + GELU -> bf16 workspace
    k_spike_gemm<<<dim3(T_ / 64, D_ / 128, B_), 256, 0, stream>>>(
        spikes, WspBf, b_spikes, day_idx, xbf);

    // windowed GEMM2 + bias + pos embedding -> out tokens 2..250
    k_proj_gemm<<<dim3((L_ + 127) / 128, H_ / 128, B_), 256, 0, stream>>>(
        xbf, WpBf, b_proj, pos_table, tstamp, out);

    // tokens 0/1, mask, timestamps
    k_epilogue<<<dim3(B_), 256, 0, stream>>>(
        block_table, day_table, spikes_mask, tstamp, block_idx, day_idx, out);
}
